// DilatedLSTM_19490561589786
// MI455X (gfx1250) — matrix operand order repaired and checked by reading, not compiled, not run on hardware
//
#include <hip/hip_runtime.h>
#include <hip/hip_bf16.h>
#include <math.h>

// ---------------------------------------------------------------------------
// Dilated LSTM:  B=8, T=2048, C=1024, H=1024, D=4  ->  32 sequences of L=512
// Phase 1: xg = xr @ W_ih^T + b   (16384 x 4096 x 1024 bf16 WMMA GEMM)
// Phase 2: persistent 512-step scan, W_hh slice resident in LDS, c in LDS,
//          one grid barrier per step to publish h (bf16, hot in L2).
// Gate columns are permuted so each WG owns {i,f,g,o} x 4 hidden units.
// ---------------------------------------------------------------------------

#define BB   8
#define TT   2048
#define CC   1024
#define HH   1024
#define DD   4
#define LL   512              // TT / DD
#define MM   (BB * DD * LL)   // 16384 rows of xg
#define G4   (4 * HH)         // 4096 gate columns
#define NWG_SCAN 256          // workgroups in scan kernel
#define UPW  4                // hidden units per scan workgroup (HH / NWG_SCAN)

typedef __bf16 v16bf __attribute__((ext_vector_type(16)));
typedef float  v8f   __attribute__((ext_vector_type(8)));
typedef unsigned int u32x4 __attribute__((ext_vector_type(4)));

union Frag16 { v16bf v; u32x4 q[2]; };

// workspace layout (bytes)
#define OFF_XB    ((size_t)0)                            // 16384*1024 bf16 = 32 MB
#define OFF_WIH   (OFF_XB  + (size_t)MM * CC * 2)        // 4096*1024 bf16  =  8 MB
#define OFF_WHH   (OFF_WIH + (size_t)G4 * CC * 2)        // 4096*1024 bf16  =  8 MB
#define OFF_BIAS  (OFF_WHH + (size_t)G4 * HH * 2)        // 4096 f32
#define OFF_HBUF  (OFF_BIAS + (size_t)G4 * 4)            // 32*1024 bf16 = 64 KB
#define OFF_BAR   (OFF_HBUF + (size_t)32 * HH * 2)       // 2 ints
#define OFF_XG    (OFF_BAR + 256)                        // 16384*4096 f32 = 256 MB

// gate-column permutation: pcol = wg*16 + gate*UPW + u  ->  orig = gate*HH + wg*UPW + u
__device__ __forceinline__ int pcol_to_orig(int pcol) {
  int wg = pcol >> 4, lg = pcol & 15;
  int g = lg >> 2, u = lg & 3;
  return g * HH + wg * UPW + u;
}

// --------------------------- conversion kernels ----------------------------

__global__ __launch_bounds__(256) void cvt_x(const float* __restrict__ x,
                                             __bf16* __restrict__ xb) {
  int idx = blockIdx.x * 256 + threadIdx.x;          // over MM*CC = 16.7M
  int m = idx >> 10, c = idx & 1023;
  int seq = m >> 9, l = m & 511;                     // L = 512
  int b = seq >> 2, d = seq & 3;
  int t = l * DD + d;                                // dilation reshuffle
  xb[(size_t)m * CC + c] = (__bf16)x[((size_t)b * TT + t) * CC + c];
}

__global__ __launch_bounds__(256) void cvt_wih(const float* __restrict__ wih,
                                               const float* __restrict__ bih,
                                               const float* __restrict__ bhh,
                                               __bf16* __restrict__ wihp,
                                               float* __restrict__ biasp) {
  int idx = blockIdx.x * 256 + threadIdx.x;          // over G4*CC
  int pcol = idx >> 10, c = idx & 1023;
  int orig = pcol_to_orig(pcol);
  wihp[(size_t)pcol * CC + c] = (__bf16)wih[(size_t)orig * CC + c];
  if (c == 0) biasp[pcol] = bih[orig] + bhh[orig];
}

__global__ __launch_bounds__(256) void cvt_whh(const float* __restrict__ whh,
                                               __bf16* __restrict__ whhp) {
  int idx = blockIdx.x * 256 + threadIdx.x;          // over G4*HH
  int pcol = idx >> 10, k = idx & 1023;
  int orig = pcol_to_orig(pcol);
  whhp[(size_t)pcol * HH + k] = (__bf16)whh[(size_t)orig * HH + k];
}

__global__ void zero_bar(int* bar) { bar[0] = 0; bar[1] = 0; }

// --------------------------- phase 1: xg GEMM ------------------------------
// Workgroup tile 64(M) x 256(N); 8 waves as 2(M) x 4(N), wave tile 32x64.
// K-loop of 32 v_wmma_f32_16x16x32_bf16 per (mt,nt).

__global__ __launch_bounds__(256) void gemm_xg(const __bf16* __restrict__ xb,
                                               const __bf16* __restrict__ wihp,
                                               const float* __restrict__ biasp,
                                               float* __restrict__ xg) {
  int bid = blockIdx.x;
  int Mbase = (bid & 255) * 64;
  int Nbase = (bid >> 8) * 256;
  int tid = threadIdx.x, lane = tid & 31, w = tid >> 5;
  int wvM = w & 1, wvN = w >> 1;
  int laneHi = lane >> 4, laneLo = lane & 15;

  v8f acc[2][4] = {};
  int rowA0 = Mbase + wvM * 32 + laneLo;             // + mt*16
  int colB0 = Nbase + wvN * 64 + laneLo;             // + nt*16

  for (int ks = 0; ks < 32; ++ks) {
    int k0 = ks * 32;
    Frag16 a[2], b[4];
#pragma unroll
    for (int mt = 0; mt < 2; ++mt) {
      const __bf16* p = xb + (size_t)(rowA0 + mt * 16) * CC + k0 + laneHi * 8;
      a[mt].q[0] = *(const u32x4*)p;                 // K = k0+kbase .. +7
      a[mt].q[1] = *(const u32x4*)(p + 16);          // K = k0+16+kbase .. +7
    }
#pragma unroll
    for (int nt = 0; nt < 4; ++nt) {
      const __bf16* p = wihp + (size_t)(colB0 + nt * 16) * CC + k0 + laneHi * 8;
      b[nt].q[0] = *(const u32x4*)p;                 // K = k0+kbase .. +7
      b[nt].q[1] = *(const u32x4*)(p + 16);          // K = k0+16+kbase .. +7
    }
#pragma unroll
    for (int mt = 0; mt < 2; ++mt)
#pragma unroll
      for (int nt = 0; nt < 4; ++nt)
        acc[mt][nt] = __builtin_amdgcn_wmma_f32_16x16x32_bf16(
            false, a[mt].v, false, b[nt].v, (short)0, acc[mt][nt], false, false);
  }

#pragma unroll
  for (int mt = 0; mt < 2; ++mt)
#pragma unroll
    for (int nt = 0; nt < 4; ++nt) {
      int col = colB0 + nt * 16;
      float bias = biasp[col];
#pragma unroll
      for (int r = 0; r < 8; ++r) {
        int row = Mbase + wvM * 32 + mt * 16 + r + laneHi * 8;
        xg[(size_t)row * G4 + col] = acc[mt][nt][r] + bias;
      }
    }
}

// --------------------------- phase 2: LSTM scan ----------------------------

__device__ __forceinline__ void grid_barrier(int* cnt, int* gen) {
  __syncthreads();
  if (threadIdx.x == 0) {
    __threadfence();
    int g = __hip_atomic_load(gen, __ATOMIC_RELAXED, __HIP_MEMORY_SCOPE_AGENT);
    int arrived = __hip_atomic_fetch_add(cnt, 1, __ATOMIC_ACQ_REL, __HIP_MEMORY_SCOPE_AGENT);
    if (arrived == NWG_SCAN - 1) {
      __hip_atomic_store(cnt, 0, __ATOMIC_RELEASE, __HIP_MEMORY_SCOPE_AGENT);
      __hip_atomic_fetch_add(gen, 1, __ATOMIC_RELEASE, __HIP_MEMORY_SCOPE_AGENT);
    } else {
      while (__hip_atomic_load(gen, __ATOMIC_ACQUIRE, __HIP_MEMORY_SCOPE_AGENT) == g) {}
    }
  }
  __syncthreads();
}

// 256 WGs x 64 threads (2 waves). WG owns 16 permuted gate columns
// (= 4 gates x 4 hidden units). W_hh slice (16 x 1024 bf16 = 32 KB) in LDS.
__global__ __launch_bounds__(64) void lstm_scan(const float* __restrict__ xg,
                                                const __bf16* __restrict__ whhp,
                                                __bf16* __restrict__ hbuf,
                                                float* __restrict__ y,
                                                int* bar) {
  __shared__ __bf16 sW[16 * HH];        // 32 KB, resident across all 512 steps
  __shared__ float  gbuf[32 * 16];      // raw h@W_hh partials for this WG
  __shared__ float  cst[32 * UPW];      // persistent cell state

  int tid = threadIdx.x, wg = blockIdx.x;
  int lane = tid & 31, wv = tid >> 5;   // wv = M-tile (rows 0-15 / 16-31)
  int laneHi = lane >> 4, laneLo = lane & 15;
  int* cnt = bar; int* gen = bar + 1;

  // preload W_hh slice: rows [wg*16, wg*16+16) of permuted bf16 matrix
  {
    const u32x4* src = (const u32x4*)(whhp + (size_t)wg * 16 * HH);
    u32x4* dst = (u32x4*)sW;
    for (int i = tid; i < 16 * HH / 8; i += 64) dst[i] = src[i];
  }
  // zero cell state and this WG's h columns
  for (int e = tid; e < 32 * UPW; e += 64) {
    cst[e] = 0.0f;
    int row = e & 31, u = e >> 5;
    hbuf[(size_t)row * HH + wg * UPW + u] = (__bf16)0.0f;
  }
  grid_barrier(cnt, gen);

  for (int l = 0; l < LL; ++l) {
    // ---- g_partial = h @ W_hh_slice^T  (32 x 16, K = 1024) ----
    v8f acc = {};
    const __bf16* hrow = hbuf + (size_t)(wv * 16 + laneLo) * HH;
#pragma unroll 4
    for (int ks = 0; ks < 32; ++ks) {
      int k0 = ks * 32;
      Frag16 a, b;
      a.q[0] = *(const u32x4*)(hrow + k0 + laneHi * 8);        // global (L2-hot)
      a.q[1] = *(const u32x4*)(hrow + k0 + 16 + laneHi * 8);
      const __bf16* wrow = sW + laneLo * HH + k0 + laneHi * 8; // LDS
      b.q[0] = *(const u32x4*)wrow;
      b.q[1] = *(const u32x4*)(wrow + 16);
      acc = __builtin_amdgcn_wmma_f32_16x16x32_bf16(
          false, a.v, false, b.v, (short)0, acc, false, false);
    }
#pragma unroll
    for (int r = 0; r < 8; ++r) {
      int row = wv * 16 + r + laneHi * 8;
      gbuf[row * 16 + laneLo] = acc[r];
    }
    __syncthreads();

    // ---- gates + state update for this WG's 4 hidden units, 32 sequences ----
    for (int e = tid; e < 32 * UPW; e += 64) {
      int row = e & 31, u = e >> 5;
      const float* xgp = xg + (size_t)(row * LL + l) * G4 + wg * 16;
      float iv = gbuf[row * 16 + 0 * UPW + u] + xgp[0 * UPW + u];
      float fv = gbuf[row * 16 + 1 * UPW + u] + xgp[1 * UPW + u];
      float gv = gbuf[row * 16 + 2 * UPW + u] + xgp[2 * UPW + u];
      float ov = gbuf[row * 16 + 3 * UPW + u] + xgp[3 * UPW + u];
      iv = 1.0f / (1.0f + __expf(-iv));
      fv = 1.0f / (1.0f + __expf(-fv));
      gv = tanhf(gv);
      ov = 1.0f / (1.0f + __expf(-ov));
      float c = fv * cst[e] + iv * gv;
      cst[e] = c;
      float h = ov * tanhf(c);
      int hid = wg * UPW + u;
      int b_ = row >> 2, d_ = row & 3;
      y[((size_t)b_ * TT + (l * DD + d_)) * HH + hid] = h;
      hbuf[(size_t)row * HH + hid] = (__bf16)h;
    }
    grid_barrier(cnt, gen);   // publish h to all WGs before next step
  }
}

// ------------------------------- launcher ----------------------------------

extern "C" void kernel_launch(void* const* d_in, const int* in_sizes, int n_in,
                              void* d_out, int out_size, void* d_ws, size_t ws_size,
                              hipStream_t stream) {
  const float* x    = (const float*)d_in[0];
  const float* wih  = (const float*)d_in[1];
  const float* whh  = (const float*)d_in[2];
  const float* bih  = (const float*)d_in[3];
  const float* bhh  = (const float*)d_in[4];

  char* ws = (char*)d_ws;
  __bf16* xb    = (__bf16*)(ws + OFF_XB);
  __bf16* wihp  = (__bf16*)(ws + OFF_WIH);
  __bf16* whhp  = (__bf16*)(ws + OFF_WHH);
  float*  biasp = (float*)(ws + OFF_BIAS);
  __bf16* hbuf  = (__bf16*)(ws + OFF_HBUF);
  int*    bar   = (int*)(ws + OFF_BAR);
  float*  xg    = (float*)(ws + OFF_XG);
  float*  y     = (float*)d_out;

  cvt_x  <<<(MM * CC) / 256, 256, 0, stream>>>(x, xb);
  cvt_wih<<<(G4 * CC) / 256, 256, 0, stream>>>(wih, bih, bhh, wihp, biasp);
  cvt_whh<<<(G4 * HH) / 256, 256, 0, stream>>>(whh, whhp);
  zero_bar<<<1, 1, 0, stream>>>(bar);

  gemm_xg<<<(MM / 64) * (G4 / 256), 256, 0, stream>>>(xb, wihp, biasp, xg);

  lstm_scan<<<NWG_SCAN, 64, 0, stream>>>(xg, whhp, hbuf, y, bar);
}
